// TransductiveGCN_19980187861405
// MI455X (gfx1250) — matrix operand order repaired and checked by reading, not compiled, not run on hardware
//
#include <hip/hip_runtime.h>
#include <hip/hip_bf16.h>
#include <cstdint>

typedef __attribute__((ext_vector_type(16))) __bf16 v16bf;
typedef __attribute__((ext_vector_type(8)))  __bf16 v8bf;
typedef __attribute__((ext_vector_type(8)))  float  v8f;
typedef __attribute__((ext_vector_type(4)))  float  v4f;

#if __has_builtin(__builtin_amdgcn_sched_barrier)
#define SCHED_FENCE() __builtin_amdgcn_sched_barrier(0)
#else
#define SCHED_FENCE()
#endif

// ---------------- utility kernels ----------------

__global__ void fill_kernel(float* __restrict__ p, float v, size_t n) {
  size_t i = (size_t)blockIdx.x * blockDim.x + threadIdx.x;
  if (i < n) p[i] = v;
}

__global__ void f2bf_kernel(const float* __restrict__ x, __bf16* __restrict__ y, size_t n) {
  size_t i = (size_t)blockIdx.x * blockDim.x + threadIdx.x;
  if (i < n) y[i] = (__bf16)x[i];
}

__global__ void degree_kernel(const long long* __restrict__ ei, float* __restrict__ deg, int E) {
  int e = blockIdx.x * blockDim.x + threadIdx.x;
  if (e < E) atomicAdd(&deg[(int)ei[(size_t)E + e]], 1.0f);
}

__global__ void rsqrt_kernel(const float* __restrict__ deg, float* __restrict__ dinv, int N) {
  int i = blockIdx.x * blockDim.x + threadIdx.x;
  if (i < N) dinv[i] = rsqrtf(deg[i]);
}

// Pack fp32 weight W[Kin x Kout] (row-major) into bf16 WMMA B-fragment tiles.
// Tile t = nt*ktiles + kt covers K rows [kt*32, kt*32+32), N cols [nt*16, nt*16+16).
// Within a tile: element index = lane*16 + v, lane = half*16 + n, k_local = half*8 + (v&7) + (v>>3)*16.
__global__ void pack_w_kernel(const float* __restrict__ W, __bf16* __restrict__ Wp,
                              int Kin, int Kout) {
  int t = blockIdx.x * blockDim.x + threadIdx.x;
  if (t >= Kin * Kout) return;
  int v    = t & 15;
  int ln   = (t >> 4) & 31;
  int tile = t >> 9;
  int idx  = ln & 15, half = ln >> 4;
  int ktiles = Kin >> 5;
  int nt = tile / ktiles, kt = tile - nt * ktiles;
  int n = nt * 16 + idx;
  int k = kt * 32 + half * 8 + (v & 7) + ((v >> 3) << 4);
  Wp[t] = (__bf16)W[(size_t)k * Kout + n];
}

// ---------------- WMMA GEMM (bf16 in, fp32 accum) ----------------
// C[N x Kout] = act(A[N x Kin] @ W + bias). A bf16 row-major, W pre-packed bf16.
// One wave computes a 16x64 output strip; ping-pong 2x-unrolled k-tile pipeline
// with sched_barrier fences so loads for slot S+1 issue before WMMAs on slot S
// (ktiles is always even here: 4, 8 or 16).
// Output: Cb (bf16) if non-null, else Cf (fp32).
// act: 0 = none, 1 = relu, 2 = leaky relu (0.1)
__global__ __launch_bounds__(256) void gemm_bf16_wmma(
    const __bf16* __restrict__ A, const __bf16* __restrict__ Wp,
    const float* __restrict__ bias, float* __restrict__ Cf, __bf16* __restrict__ Cb,
    int N, int Kin, int Kout, int act)
{
  const int lane = threadIdx.x & 31;
  const int wave = threadIdx.x >> 5;
  const int strips = Kout >> 6;             // 64-column strips
  const int numRowTiles = N >> 4;
  const int task = blockIdx.x * 8 + wave;   // wave-uniform guard -> EXEC all-ones for WMMA
  if (task >= numRowTiles * strips) return;
  const int tsk = __builtin_amdgcn_readfirstlane(task);   // provably wave-uniform
  const int rt = tsk / strips;
  const int cs = tsk - rt * strips;
  const unsigned idx = lane & 15, half = lane >> 4;
  const int ktiles = Kin >> 5;

  // Uniform base pointers + divergent unsigned 32-bit offsets (zext, no ashr).
  const __bf16* au = A + (size_t)rt * (16 * Kin);
  const unsigned aoff = idx * (unsigned)Kin + half * 8;   // lane-local A offset (elements)
  const __bf16* bu = Wp + (((size_t)(cs * 4) * ktiles) << 9);
  const unsigned boff = (unsigned)lane << 4;              // lane-local B offset (elements)

  v8f acc[4] = {};

  v8bf a0[2], a1[2];
  v16bf b[2][4];

  // prologue: slot 0 <- k-tile 0
  a0[0] = *(const v8bf*)(au + aoff);
  a1[0] = *(const v8bf*)(au + (aoff + 16u));
#pragma unroll
  for (int j = 0; j < 4; ++j)
    b[0][j] = *(const v16bf*)(bu + (boff + (unsigned)((j * ktiles) << 9)));

  for (int kt = 0; kt < ktiles; kt += 2) {
    // issue slot 1 <- k-tile kt+1 (always valid: ktiles even)
    const unsigned k1 = (unsigned)(kt + 1);
    a0[1] = *(const v8bf*)(au + (aoff + k1 * 32u));
    a1[1] = *(const v8bf*)(au + (aoff + k1 * 32u + 16u));
#pragma unroll
    for (int j = 0; j < 4; ++j)
      b[1][j] = *(const v16bf*)(bu + (boff + (unsigned)((j * ktiles + (int)k1) << 9)));
    __builtin_prefetch(au + (aoff + (unsigned)(kt + 4) * 32u), 0, 1); // global_prefetch_b8
    SCHED_FENCE();   // keep slot-1 loads issued above the slot-0 WMMAs

    // WMMA on slot 0 (drains only down to slot-1's in-flight loads)
    {
      v16bf af = __builtin_shufflevector(a0[0], a1[0],
                                         0, 1, 2, 3, 4, 5, 6, 7,
                                         8, 9, 10, 11, 12, 13, 14, 15);
#pragma unroll
      for (int j = 0; j < 4; ++j)
        acc[j] = __builtin_amdgcn_wmma_f32_16x16x32_bf16(
            false, af, false, b[0][j], (short)0, acc[j], false, false);
    }
    SCHED_FENCE();

    // issue slot 0 <- k-tile kt+2 (clamped; redundant reload on final trip)
    const unsigned k2 = (unsigned)((kt + 2 < ktiles) ? kt + 2 : kt);
    a0[0] = *(const v8bf*)(au + (aoff + k2 * 32u));
    a1[0] = *(const v8bf*)(au + (aoff + k2 * 32u + 16u));
#pragma unroll
    for (int j = 0; j < 4; ++j)
      b[0][j] = *(const v16bf*)(bu + (boff + (unsigned)((j * ktiles + (int)k2) << 9)));
    SCHED_FENCE();   // keep slot-0 loads issued above the slot-1 WMMAs

    // WMMA on slot 1
    {
      v16bf af = __builtin_shufflevector(a0[1], a1[1],
                                         0, 1, 2, 3, 4, 5, 6, 7,
                                         8, 9, 10, 11, 12, 13, 14, 15);
#pragma unroll
      for (int j = 0; j < 4; ++j)
        acc[j] = __builtin_amdgcn_wmma_f32_16x16x32_bf16(
            false, af, false, b[1][j], (short)0, acc[j], false, false);
    }
    SCHED_FENCE();
  }

  // epilogue: uniform row-tile base + int offsets
  float*  cfu = Cf ? Cf + (size_t)rt * (16 * Kout) : nullptr;
  __bf16* cbu = Cb ? Cb + (size_t)rt * (16 * Kout) : nullptr;
#pragma unroll
  for (int j = 0; j < 4; ++j) {
    const int col = cs * 64 + j * 16 + (int)idx;
    const float bv = bias ? bias[col] : 0.0f;
#pragma unroll
    for (int r = 0; r < 8; ++r) {
      float v = acc[j][r] + bv;
      if (act == 1)      v = fmaxf(v, 0.0f);
      else if (act == 2) v = v > 0.0f ? v : 0.1f * v;
      const unsigned o = (half * 8u + (unsigned)r) * (unsigned)Kout + (unsigned)col;
      if (cbu) cbu[o] = (__bf16)v;
      else     cfu[o] = v;
    }
  }
}

// ---------------- LayerNorm over 128 features, one wave per row, bf16 out ----------------
__global__ __launch_bounds__(256) void ln_kernel(const float* __restrict__ X,
                                                 const float* __restrict__ g,
                                                 const float* __restrict__ b,
                                                 __bf16* __restrict__ Y, int N) {
  int wave = threadIdx.x >> 5, lane = threadIdx.x & 31;
  int row = blockIdx.x * 8 + wave;
  if (row >= N) return;
  const float* xr = X + (size_t)row * 128;
  float v[4];
#pragma unroll
  for (int i = 0; i < 4; ++i) v[i] = xr[lane + 32 * i];
  float s = v[0] + v[1] + v[2] + v[3];
#pragma unroll
  for (int m = 16; m >= 1; m >>= 1) s += __shfl_xor(s, m, 32);
  float mean = s * (1.0f / 128.0f);
  float q = 0.0f;
#pragma unroll
  for (int i = 0; i < 4; ++i) { float d = v[i] - mean; q += d * d; }
#pragma unroll
  for (int m = 16; m >= 1; m >>= 1) q += __shfl_xor(q, m, 32);
  float rstd = rsqrtf(q * (1.0f / 128.0f) + 1e-5f);
  __bf16* yr = Y + (size_t)row * 128;
#pragma unroll
  for (int i = 0; i < 4; ++i) {
    int f = lane + 32 * i;
    yr[f] = (__bf16)((v[i] - mean) * rstd * g[f] + b[f]);
  }
}

// ---------------- edge scatter: agg[col] += dinv[row]*dinv[col] * xw[row] ----------------
__global__ __launch_bounds__(256) void scatter_kernel(const long long* __restrict__ ei,
                                                      const float* __restrict__ dinv,
                                                      const float* __restrict__ xw,
                                                      float* __restrict__ agg,
                                                      int E, int ho) {
  long long tid = (long long)blockIdx.x * blockDim.x + threadIdx.x;
  int chunks = ho >> 2;
  if (tid >= (long long)E * chunks) return;
  int e = (int)(tid / chunks);
  int c = (int)(tid - (long long)e * chunks);
  int r   = (int)ei[e];
  int col = (int)ei[(size_t)E + e];
  float nrm = dinv[r] * dinv[col];
  v4f xv = *(const v4f*)(xw + (size_t)r * ho + c * 4);
  float* ap = agg + (size_t)col * ho + c * 4;
  atomicAdd(ap + 0, nrm * xv[0]);
  atomicAdd(ap + 1, nrm * xv[1]);
  atomicAdd(ap + 2, nrm * xv[2]);
  atomicAdd(ap + 3, nrm * xv[3]);
}

// ---------------- combine: h = elu01(agg + self_norm*xw + cb + skip), bf16 out ----------------
__global__ void combine_kernel(const float* __restrict__ agg, const float* __restrict__ xw,
                               const float* __restrict__ s,   const float* __restrict__ cb,
                               const float* __restrict__ dinv, __bf16* __restrict__ hout,
                               int N, int ho) {
  size_t i = (size_t)blockIdx.x * blockDim.x + threadIdx.x;
  if (i >= (size_t)N * ho) return;
  int node = (int)(i / ho);
  int f = (int)(i - (size_t)node * ho);
  float di = dinv[node];
  float vv = agg[i] + 2.0f * di * di * xw[i] + cb[f] + s[i];
  hout[i] = (__bf16)(vv > 0.0f ? vv : 0.1f * expm1f(vv));
}

// ---------------- decoder final: out = h[:,0:256] . w + b, one wave per node ----------------
__global__ __launch_bounds__(256) void dec2_kernel(const float* __restrict__ h,
                                                   const float* __restrict__ w,
                                                   const float* __restrict__ b,
                                                   float* __restrict__ out, int N) {
  int wave = threadIdx.x >> 5, lane = threadIdx.x & 31;
  int node = blockIdx.x * 8 + wave;
  if (node >= N) return;
  const float* hp = h + (size_t)node * 256;
  float s = 0.0f;
#pragma unroll
  for (int i = 0; i < 8; ++i) s += hp[lane + 32 * i] * w[lane + 32 * i];
#pragma unroll
  for (int m = 16; m >= 1; m >>= 1) s += __shfl_xor(s, m, 32);
  if (lane == 0) out[node] = s + b[0];
}

// ---------------- host orchestration ----------------

extern "C" void kernel_launch(void* const* d_in, const int* in_sizes, int n_in,
                              void* d_out, int out_size, void* d_ws, size_t ws_size,
                              hipStream_t stream) {
  const float*     x      = (const float*)d_in[0];
  const long long* ei     = (const long long*)d_in[1];
  const float* enc_w1 = (const float*)d_in[2];  const float* enc_b1 = (const float*)d_in[3];
  const float* enc_w2 = (const float*)d_in[4];  const float* enc_b2 = (const float*)d_in[5];
  const float* ln_g   = (const float*)d_in[6];  const float* ln_b   = (const float*)d_in[7];
  const float* skw[3] = {(const float*)d_in[8],  (const float*)d_in[12], (const float*)d_in[16]};
  const float* skb[3] = {(const float*)d_in[9],  (const float*)d_in[13], (const float*)d_in[17]};
  const float* cvw[3] = {(const float*)d_in[10], (const float*)d_in[14], (const float*)d_in[18]};
  const float* cvb[3] = {(const float*)d_in[11], (const float*)d_in[15], (const float*)d_in[19]};
  const float* dec_w1 = (const float*)d_in[20]; const float* dec_b1 = (const float*)d_in[21];
  const float* dec_w2 = (const float*)d_in[22]; const float* dec_b2 = (const float*)d_in[23];

  const int N = in_sizes[0] / 128;   // 50000, multiple of 16
  const int E = in_sizes[1] / 2;     // 800000
  const int DIMS[4] = {128, 256, 512, 256};

  char* base = (char*)d_ws;
  size_t cur = 0;
  auto alloc = [&](size_t bytes) -> char* {
    char* p = base + cur;
    cur += (bytes + 255) & ~(size_t)255;
    return p;
  };
  float*  deg   = (float*)alloc((size_t)N * 4);
  float*  dinv  = (float*)alloc((size_t)N * 4);
  __bf16* wp_e1 = (__bf16*)alloc((size_t)128 * 128 * 2);
  __bf16* wp_e2 = (__bf16*)alloc((size_t)128 * 128 * 2);
  __bf16* wp_sk[3]; __bf16* wp_cv[3];
  for (int i = 0; i < 3; ++i) {
    wp_sk[i] = (__bf16*)alloc((size_t)DIMS[i] * DIMS[i + 1] * 2);
    wp_cv[i] = (__bf16*)alloc((size_t)DIMS[i] * DIMS[i + 1] * 2);
  }
  __bf16* wp_d1 = (__bf16*)alloc((size_t)256 * 256 * 2);
  __bf16* xbf   = (__bf16*)alloc((size_t)N * 128 * 2);  // input features, bf16
  __bf16* hbf   = (__bf16*)alloc((size_t)N * 512 * 2);  // activations, bf16
  float* buf1 = (float*)alloc((size_t)N * 512 * 4);     // xw / tmp (fp32)
  float* buf2 = (float*)alloc((size_t)N * 512 * 4);     // skip / tmp (fp32)
  float* buf3 = (float*)alloc((size_t)N * 512 * 4);     // agg (fp32)
  (void)ws_size; (void)n_in; (void)out_size;

  auto grid1 = [](size_t n) { return (unsigned)((n + 255) / 256); };

  // normalization terms
  fill_kernel<<<grid1((size_t)N), 256, 0, stream>>>(deg, 2.0f, (size_t)N);
  degree_kernel<<<grid1((size_t)E), 256, 0, stream>>>(ei, deg, E);
  rsqrt_kernel<<<grid1((size_t)N), 256, 0, stream>>>(deg, dinv, N);

  // convert input features once
  f2bf_kernel<<<grid1((size_t)N * 128), 256, 0, stream>>>(x, xbf, (size_t)N * 128);

  // pack all weights to bf16 B-fragment layout
  auto pack = [&](const float* W, __bf16* Wp, int ki, int ko) {
    pack_w_kernel<<<grid1((size_t)ki * ko), 256, 0, stream>>>(W, Wp, ki, ko);
  };
  pack(enc_w1, wp_e1, 128, 128);
  pack(enc_w2, wp_e2, 128, 128);
  for (int i = 0; i < 3; ++i) {
    pack(skw[i], wp_sk[i], DIMS[i], DIMS[i + 1]);
    pack(cvw[i], wp_cv[i], DIMS[i], DIMS[i + 1]);
  }
  pack(dec_w1, wp_d1, 256, 256);

  auto gemm = [&](const __bf16* A, const __bf16* Wp, const float* bias,
                  float* Cf, __bf16* Cb, int ki, int ko, int act) {
    int tasks = (N >> 4) * (ko >> 6);
    gemm_bf16_wmma<<<(tasks + 7) / 8, 256, 0, stream>>>(A, Wp, bias, Cf, Cb, N, ki, ko, act);
  };

  // encoder: relu(x@w1+b1) (bf16 out) -> @w2+b2 (fp32 out) -> LayerNorm (bf16 h)
  gemm(xbf, wp_e1, enc_b1, nullptr, hbf, 128, 128, 1);
  gemm(hbf, wp_e2, enc_b2, buf2, nullptr, 128, 128, 0);
  ln_kernel<<<(unsigned)((N + 7) / 8), 256, 0, stream>>>(buf2, ln_g, ln_b, hbf, N);

  // GCN blocks
  for (int i = 0; i < 3; ++i) {
    int hi = DIMS[i], ho = DIMS[i + 1];
    gemm(hbf, wp_cv[i], nullptr, buf1, nullptr, hi, ho, 0);   // xw = h @ conv_w
    gemm(hbf, wp_sk[i], skb[i],  buf2, nullptr, hi, ho, 0);   // skip = h @ skip_w + b
    fill_kernel<<<grid1((size_t)N * ho), 256, 0, stream>>>(buf3, 0.0f, (size_t)N * ho);
    scatter_kernel<<<grid1((size_t)E * (ho >> 2)), 256, 0, stream>>>(ei, dinv, buf1, buf3, E, ho);
    combine_kernel<<<grid1((size_t)N * ho), 256, 0, stream>>>(buf3, buf1, buf2, cvb[i], dinv,
                                                              hbf, N, ho);
  }

  // decoder
  gemm(hbf, wp_d1, dec_b1, buf1, nullptr, 256, 256, 2);       // leaky relu 0.1
  dec2_kernel<<<(unsigned)((N + 7) / 8), 256, 0, stream>>>(buf1, dec_w2, dec_b2,
                                                           (float*)d_out, N);
}
